// BidirectionalSimilarityAttentionFusion__51256139710851
// MI455X (gfx1250) — matrix operand order repaired and checked by reading, not compiled, not run on hardware
//
#include <hip/hip_runtime.h>
#include <stdint.h>

// ---- CDNA5 wave32 WMMA types ----
typedef __attribute__((ext_vector_type(16))) __bf16       v16bf;
typedef __attribute__((ext_vector_type(8)))  float        v8f;
typedef __attribute__((ext_vector_type(4)))  float        v4f;
typedef __attribute__((ext_vector_type(4)))  unsigned int v4u;

union FragB { v16bf v; v4u u[2]; };   // 32 B: one WMMA A/B operand (16 bf16)

#define Dm   768
#define TQ   16        // query rows per q-group
#define TK   64        // key tile
#define NGRP 2         // q-groups (tiles) per workgroup
#define NTH  256       // 8 waves: waves 0-3 -> group 0, waves 4-7 -> group 1

// padded LDS strides: conflict-free on the 64-bank LDS
#define SQP  776   // lane step = 4 dwords  -> banks 4n (perfect b128 tiling)
#define SKP  776
#define SKTP 72    // lane step = 36 dwords -> 16 distinct 4-bank groups
#define SSP  68    // lane step = 4 dwords
#define SPP  72

__device__ __forceinline__ unsigned pack2bf(float a, float b) {
  union { __bf16 h[2]; unsigned u; } p;
  p.h[0] = (__bf16)a; p.h[1] = (__bf16)b;           // hardware v_cvt bf16 path
  return p.u;
}
__device__ __forceinline__ unsigned short f2bf(float x) {
  union { __bf16 h; unsigned short s; } p;
  p.h = (__bf16)x; return p.s;
}

// gfx1250 async DMA: global -> LDS, 16 bytes per lane, tracked by ASYNCcnt
__device__ __forceinline__ void async_b128(const void* g, const void* lds) {
  unsigned long long ga = (unsigned long long)(uintptr_t)g;
  unsigned           la = (unsigned)(uintptr_t)lds;   // generic LDS ptr: offset in [31:0]
  asm volatile("global_load_async_to_lds_b128 %0, %1, off"
               :: "v"(la), "v"(ga) : "memory");
}
__device__ __forceinline__ void wait_async0() {
#if __has_builtin(__builtin_amdgcn_s_wait_asynccnt)
  __builtin_amdgcn_s_wait_asynccnt(0);
#else
  asm volatile("s_wait_asynccnt 0x0" ::: "memory");
#endif
}

// ---- pre-pass 1: f32 -> bf16 (row-major), memory-bound one-shot ----
__global__ __launch_bounds__(256)
void cvt_bf16_kernel(const float* __restrict__ src, unsigned short* __restrict__ dst) {
  size_t i = (size_t)blockIdx.x * blockDim.x + threadIdx.x;   // one float4 each
  v4f v = *reinterpret_cast<const v4f*>(src + i * 4);
  *reinterpret_cast<unsigned*>(dst + i * 4)     = pack2bf(v.x, v.y);
  *reinterpret_cast<unsigned*>(dst + i * 4 + 2) = pack2bf(v.z, v.w);
}

// ---- pre-pass 2: bf16 [B][N][Dm] -> [B][Dm][N], LDS-tiled (coalesced both sides) ----
__global__ __launch_bounds__(256)
void transpose_bf16_kernel(const unsigned short* __restrict__ src,
                           unsigned short* __restrict__ dst, int N) {
  __shared__ unsigned short t[64][72];
  const int nb = blockIdx.x * 64, db = blockIdx.y * 64, b = blockIdx.z;
  const unsigned short* s = src + (size_t)b * N * Dm;
  unsigned short*       d = dst + (size_t)b * Dm * N;
  for (int i = threadIdx.x; i < 64 * 16; i += 256) {
    int r = i >> 4, c4 = (i & 15) * 4;
    const unsigned short* p = s + (size_t)(nb + r) * Dm + db + c4;
    t[r][c4] = p[0]; t[r][c4 + 1] = p[1]; t[r][c4 + 2] = p[2]; t[r][c4 + 3] = p[3];
  }
  __syncthreads();
  for (int i = threadIdx.x; i < 64 * 16; i += 256) {
    int dd = i >> 4, n4 = (i & 15) * 4;
    unsigned short* q = d + (size_t)(db + dd) * N + nb + n4;
    q[0] = t[n4][dd]; q[1] = t[n4 + 1][dd]; q[2] = t[n4 + 2][dd]; q[3] = t[n4 + 3][dd];
  }
}

// ---- flash-attention pass: Out rows [outRowOff, outRowOff+nq) = softmax(Q K^T) K ----
// 8 waves, 2 q-groups of 16 rows sharing one K/V tile (amortizes the tile DMA,
// 2 waves/SIMD so one group's WMMA hides the other's softmax / async wait).
// FAST: Qp/Kp bf16 row-major, KTp bf16 transposed [Dm][nkv]; tiles DMA'd via async.
// !FAST: Qp/Kp f32; convert + transpose inside the loader (no workspace needed).
template<bool FAST>
__global__ __launch_bounds__(NTH)
void xattn_kernel(const void* __restrict__ Qp, const void* __restrict__ Kp,
                  const void* __restrict__ KTp, float* __restrict__ Out,
                  int nq, int nkv, int outRowOff)
{
  __shared__ unsigned short sQ [NGRP][TQ * SQP];   // Q tiles  bf16   (~49.7 KB)
  __shared__ unsigned short sK [TK * SKP];         // K tile   bf16   (~97  KB)
  __shared__ unsigned short sKT[Dm * SKTP];        // V^T tile bf16   (~108 KB)
  __shared__ float          sS [NGRP][TQ * SSP];   // raw scores f32
  __shared__ unsigned short sP [NGRP][TQ * SPP];   // softmax numerators bf16
  __shared__ float sScale[NGRP][TQ];
  __shared__ float sInvL [NGRP][TQ];

  const int tid  = threadIdx.x;
  const int g    = tid >> 7;          // q-group 0/1
  const int wq   = (tid >> 5) & 3;    // wave role within group
  const int lane = tid & 31;
  const int col  = lane & 15;
  const int hi   = lane >> 4;
  const int row  = col;
  const int b    = blockIdx.y;
  const int qbase = (blockIdx.x * NGRP + g) * TQ;

  unsigned short* mQ = sQ[g];
  float*          mS = sS[g];
  unsigned short* mP = sP[g];

  // ---- Q tiles (both groups loaded cooperatively) ----
  if (FAST) {
    const unsigned short* Qb = (const unsigned short*)Qp + ((size_t)b * nq) * Dm;
    for (int i = tid; i < NGRP * TQ * (Dm / 8); i += NTH) {
      int gg = i / (TQ * (Dm / 8)), j = i % (TQ * (Dm / 8));
      int r = j / (Dm / 8), c = j % (Dm / 8);
      int qr = (blockIdx.x * NGRP + gg) * TQ + r;
      async_b128(Qb + (size_t)qr * Dm + c * 8, &sQ[gg][r * SQP + c * 8]);
    }            // waited together with tile-0 asyncs below
  } else {
    const float* Qf = (const float*)Qp + ((size_t)b * nq) * Dm;
    for (int i = tid; i < NGRP * TQ * (Dm / 4); i += NTH) {
      int gg = i / (TQ * (Dm / 4)), j = i % (TQ * (Dm / 4));
      int r = j / (Dm / 4), c4 = (j % (Dm / 4)) * 4;
      int qr = (blockIdx.x * NGRP + gg) * TQ + r;
      v4f v = *reinterpret_cast<const v4f*>(Qf + (size_t)qr * Dm + c4);
      *reinterpret_cast<unsigned*>(&sQ[gg][r * SQP + c4])     = pack2bf(v.x, v.y);
      *reinterpret_cast<unsigned*>(&sQ[gg][r * SQP + c4 + 2]) = pack2bf(v.z, v.w);
    }
  }

  v8f Ov[12];
  #pragma unroll
  for (int s = 0; s < 12; ++s) Ov[s] = {};
  float m_run = -1e30f, l_run = 0.0f;
  const int ntile = nkv / TK;

  for (int kt = 0; kt < ntile; ++kt) {
    __syncthreads();   // prior PV reads finished before tile overwrite

    if (FAST) {
      const unsigned short* Kb  = (const unsigned short*)Kp  + ((size_t)b * nkv + kt * TK) * Dm;
      const unsigned short* KTb = (const unsigned short*)KTp + (size_t)b * Dm * nkv + (size_t)kt * TK;
      for (int i = tid; i < TK * (Dm / 8); i += NTH) {        // row-major K tile
        int r = i / (Dm / 8), c = i % (Dm / 8);
        async_b128(Kb + (size_t)r * Dm + c * 8, &sK[r * SKP + c * 8]);
      }
      for (int i = tid; i < Dm * (TK / 8); i += NTH) {        // V^T tile
        int d = i >> 3, c = i & 7;
        async_b128(KTb + (size_t)d * nkv + c * 8, &sKT[d * SKTP + c * 8]);
      }
      wait_async0();
    } else {
      const float* Kf = (const float*)Kp + ((size_t)b * nkv + kt * TK) * Dm;
      if (kt + 1 < ntile)
        __builtin_prefetch((const float*)Kp + ((size_t)b * nkv + (kt + 1) * TK) * Dm + tid * 96, 0, 0);
      for (int i = tid; i < TK * (Dm / 4); i += NTH) {
        int r = i / (Dm / 4), c4 = (i % (Dm / 4)) * 4;
        v4f v = *reinterpret_cast<const v4f*>(Kf + (size_t)r * Dm + c4);
        unsigned short b0 = f2bf(v.x), b1 = f2bf(v.y), b2 = f2bf(v.z), b3 = f2bf(v.w);
        *reinterpret_cast<unsigned*>(&sK[r * SKP + c4])     = (unsigned)b0 | ((unsigned)b1 << 16);
        *reinterpret_cast<unsigned*>(&sK[r * SKP + c4 + 2]) = (unsigned)b2 | ((unsigned)b3 << 16);
        sKT[(c4    ) * SKTP + r] = b0;
        sKT[(c4 + 1) * SKTP + r] = b1;
        sKT[(c4 + 2) * SKTP + r] = b2;
        sKT[(c4 + 3) * SKTP + r] = b3;
      }
    }
    __syncthreads();

    // ---- S = Q K^T : wave wq -> 16x16 subtile, key cols [16wq, 16wq+16) ----
    v8f acc = {};
    #pragma unroll
    for (int kc = 0; kc < Dm / 32; ++kc) {
      FragB a, bm;
      int ao = col * SQP + kc * 32 + hi * 8;
      a.u[0] = *reinterpret_cast<const v4u*>(&mQ[ao]);
      a.u[1] = *reinterpret_cast<const v4u*>(&mQ[ao + 16]);
      int bo = (wq * 16 + col) * SKP + kc * 32 + hi * 8;
      bm.u[0] = *reinterpret_cast<const v4u*>(&sK[bo]);
      bm.u[1] = *reinterpret_cast<const v4u*>(&sK[bo + 16]);
      acc = __builtin_amdgcn_wmma_f32_16x16x32_bf16(false, a.v, false, bm.v,
                                                    (short)0, acc, false, false);
    }
    #pragma unroll
    for (int r = 0; r < 8; ++r)
      mS[(r + hi * 8) * SSP + wq * 16 + col] = acc[r];
    __syncthreads();

    // ---- online softmax (lane owns one row; redundant across the 4 group waves) ----
    float mloc = -1e30f;
    #pragma unroll
    for (int j = 0; j < TK / 4; ++j) {
      v4f sv = *reinterpret_cast<const v4f*>(&mS[row * SSP + j * 4]);
      mloc = fmaxf(mloc, fmaxf(fmaxf(sv.x, sv.y), fmaxf(sv.z, sv.w)));
    }
    float m_new = fmaxf(m_run, mloc);
    float scl   = __expf(m_run - m_new);
    float sum   = 0.0f;
    #pragma unroll
    for (int j = 0; j < TK / 4; ++j) {
      v4f sv = *reinterpret_cast<const v4f*>(&mS[row * SSP + j * 4]);
      float e0 = __expf(sv.x - m_new), e1 = __expf(sv.y - m_new);
      float e2 = __expf(sv.z - m_new), e3 = __expf(sv.w - m_new);
      sum += (e0 + e1) + (e2 + e3);
      if (hi == 0 && (j >> 2) == wq) {       // wave wq writes P cols [16wq,16wq+16)
        *reinterpret_cast<unsigned*>(&mP[row * SPP + j * 4])     = pack2bf(e0, e1);
        *reinterpret_cast<unsigned*>(&mP[row * SPP + j * 4 + 2]) = pack2bf(e2, e3);
      }
    }
    l_run = l_run * scl + sum;
    m_run = m_new;
    if (wq == 0 && hi == 0) sScale[g][row] = scl;
    __syncthreads();

    // ---- rescale O, then O += P V ; wave wq owns D cols [192wq, 192wq+192) ----
    float sc[8];
    #pragma unroll
    for (int r = 0; r < 8; ++r) sc[r] = sScale[g][r + hi * 8];
    #pragma unroll
    for (int s = 0; s < 12; ++s) {
      #pragma unroll
      for (int r = 0; r < 8; ++r) Ov[s][r] *= sc[r];
    }
    #pragma unroll
    for (int kk = 0; kk < 2; ++kk) {
      FragB a;
      int ao = col * SPP + kk * 32 + hi * 8;
      a.u[0] = *reinterpret_cast<const v4u*>(&mP[ao]);
      a.u[1] = *reinterpret_cast<const v4u*>(&mP[ao + 16]);
      #pragma unroll
      for (int s = 0; s < 12; ++s) {
        FragB bm;
        int bo = (wq * 192 + s * 16 + col) * SKTP + kk * 32 + hi * 8;
        bm.u[0] = *reinterpret_cast<const v4u*>(&sKT[bo]);
        bm.u[1] = *reinterpret_cast<const v4u*>(&sKT[bo + 16]);
        Ov[s] = __builtin_amdgcn_wmma_f32_16x16x32_bf16(false, a.v, false, bm.v,
                                                        (short)0, Ov[s], false, false);
      }
    }
  }

  // ---- finalize ----
  __syncthreads();
  if (wq == 0 && hi == 0) sInvL[g][row] = 1.0f / l_run;
  __syncthreads();
  float inv[8];
  #pragma unroll
  for (int r = 0; r < 8; ++r) inv[r] = sInvL[g][r + hi * 8];

  float* Ob = Out + ((size_t)b * 3072 + outRowOff + qbase) * Dm;
  #pragma unroll
  for (int s = 0; s < 12; ++s) {
    int d = wq * 192 + s * 16 + col;
    #pragma unroll
    for (int r = 0; r < 8; ++r)
      Ob[(size_t)(r + hi * 8) * Dm + d] = Ov[s][r] * inv[r];
  }
}

extern "C" void kernel_launch(void* const* d_in, const int* in_sizes, int n_in,
                              void* d_out, int out_size, void* d_ws, size_t ws_size,
                              hipStream_t stream) {
  (void)in_sizes; (void)n_in; (void)out_size;
  const float* g   = (const float*)d_in[0];   // [32,1024,768] f32
  const float* l   = (const float*)d_in[1];   // [32,2048,768] f32
  float*       out = (float*)d_out;           // [32,3072,768] f32
  const int B = 32, Ng = 1024, Nl = 2048;
  const size_t eG = (size_t)B * Ng * Dm;
  const size_t eL = (size_t)B * Nl * Dm;
  const size_t need = 2 * (eG + eL) * sizeof(unsigned short);   // ~302 MB

  if (ws_size >= need) {
    unsigned short* wG  = (unsigned short*)d_ws;     // bf16 G   [B][Ng][Dm]
    unsigned short* wL  = wG + eG;                   // bf16 L   [B][Nl][Dm]
    unsigned short* wGT = wL + eL;                   // bf16 G^T [B][Dm][Ng]
    unsigned short* wLT = wGT + eG;                  // bf16 L^T [B][Dm][Nl]

    cvt_bf16_kernel<<<dim3((unsigned)(eG / 4 / 256)), dim3(256), 0, stream>>>(g, wG);
    cvt_bf16_kernel<<<dim3((unsigned)(eL / 4 / 256)), dim3(256), 0, stream>>>(l, wL);
    transpose_bf16_kernel<<<dim3(Ng / 64, Dm / 64, B), dim3(256), 0, stream>>>(wG, wGT, Ng);
    transpose_bf16_kernel<<<dim3(Nl / 64, Dm / 64, B), dim3(256), 0, stream>>>(wL, wLT, Nl);

    // pass 1: attended_local  = softmax(G L^T) L -> rows [0, Ng)
    xattn_kernel<true><<<dim3(Ng / (NGRP * TQ), B), dim3(NTH), 0, stream>>>(wG, wL, wLT, out, Ng, Nl, 0);
    // pass 2: attended_global = softmax(L G^T) G -> rows [Ng, Ng+Nl)
    xattn_kernel<true><<<dim3(Nl / (NGRP * TQ), B), dim3(NTH), 0, stream>>>(wL, wG, wGT, out, Nl, Ng, Ng);
  } else {
    xattn_kernel<false><<<dim3(Ng / (NGRP * TQ), B), dim3(NTH), 0, stream>>>(g, l, nullptr, out, Ng, Nl, 0);
    xattn_kernel<false><<<dim3(Nl / (NGRP * TQ), B), dim3(NTH), 0, stream>>>(l, g, nullptr, out, Nl, Ng, Ng);
  }
}
